// MultiHeadSelfAttention_18605798326362
// MI455X (gfx1250) — matrix operand order repaired and checked by reading, not compiled, not run on hardware
//
#include <hip/hip_runtime.h>

// ---------------------------------------------------------------------------
// Multi-head self-attention for MI455X (gfx1250), bf16 WMMA + f32 accumulate.
// B=2, S=2048, D=1024, H=16, dk=64.  scores scaled by 1/sqrt(64) = 0.125.
// LDS-staged tiles via CDNA5 async global->LDS loads, double buffered.
// ---------------------------------------------------------------------------

typedef __bf16 bf16;
typedef bf16  v16bf __attribute__((ext_vector_type(16)));
typedef bf16  v8bf  __attribute__((ext_vector_type(8)));
typedef bf16  v4bf  __attribute__((ext_vector_type(4)));
typedef float v8f   __attribute__((ext_vector_type(8)));

#define BATCH   2
#define SEQ     2048
#define DMODEL  1024
#define NHEAD   16
#define DK      64

#define LDS_AS __attribute__((address_space(3)))

#if __has_builtin(__builtin_amdgcn_s_wait_asynccnt)
#define WAIT_ASYNC(n) __builtin_amdgcn_s_wait_asynccnt(n)
#elif defined(__AMDGCN__)
#define WAIT_ASYNC(n) asm volatile("s_wait_asynccnt %0" ::"i"(n) : "memory")
#else
#define WAIT_ASYNC(n)
#endif

// 32-bit LDS byte offset of a generic pointer into shared memory
__device__ __forceinline__ unsigned lds_addr_of(const void* p) {
    return (unsigned)(unsigned long long)(LDS_AS const void*)p;
}

// async copy of 16B / 32B (per lane) from global to LDS, ASYNCcnt-tracked.
// ISA 15.18.3: GLOBAL_LOAD_ASYNC_TO_LDS_B128  vdst(=LDS addr), vaddr64, off
__device__ __forceinline__ void async_cp16(const bf16* g, bf16* l) {
#if defined(__AMDGCN__)
    asm volatile("global_load_async_to_lds_b128 %0, %1, off"
                 :: "v"(lds_addr_of(l)), "v"(g) : "memory");
#else
    *(v8bf*)l = *(const v8bf*)g;
#endif
}
__device__ __forceinline__ void async_cp32(const bf16* g, bf16* l) {
#if defined(__AMDGCN__)
    unsigned la = lds_addr_of(l);
    asm volatile("global_load_async_to_lds_b128 %0, %1, off"
                 :: "v"(la), "v"(g) : "memory");
    asm volatile("global_load_async_to_lds_b128 %0, %1, off offset:16"
                 :: "v"(la), "v"(g) : "memory");
#else
    *(v8bf*)l       = *(const v8bf*)g;
    *(v8bf*)(l + 8) = *(const v8bf*)(g + 8);
#endif
}

__device__ __forceinline__ bf16 to_bf16(float f) {
    unsigned u = __float_as_uint(f);
    u += 0x7FFFu + ((u >> 16) & 1u);          // round-to-nearest-even
    unsigned short h = (unsigned short)(u >> 16);
    bf16 r;
    __builtin_memcpy(&r, &h, 2);
    return r;
}

__device__ __forceinline__ v8f wmma_bf16(v16bf a, v16bf b, v8f c) {
    return __builtin_amdgcn_wmma_f32_16x16x32_bf16(false, a, false, b,
                                                   (short)0, c, false, false);
}

// A-fragment (16x32, bf16): lane = {half, r}; per-lane row m = r,
// 16 contiguous K at k0 + half*16 (two 16B loads).
__device__ __forceinline__ v16bf load_a_frag(const bf16* __restrict__ base,
                                             int ld, int row0, int k0, int lane) {
    int r = lane & 15, half = lane >> 4;
    const bf16* p = base + (size_t)(row0 + r) * ld + k0 + half * 16;
    v8bf lo = *(const v8bf*)p;
    v8bf hi = *(const v8bf*)(p + 8);
    v16bf out;
#pragma unroll
    for (int i = 0; i < 8; ++i) { out[i] = lo[i]; out[i + 8] = hi[i]; }
    return out;
}

// B-fragment (32x16, bf16): lane owns column n = ncol0 + (lane&15),
// holds 16 contiguous K values starting at k0 + half*16 (one 32B read).
// base is row-major [n][k] (i.e. B transposed in memory), stride ld.
__device__ __forceinline__ v16bf load_b_frag(const bf16* __restrict__ base,
                                             int ld, int ncol0, int k0, int lane) {
    int c = lane & 15, half = lane >> 4;
    return *(const v16bf*)(base + (size_t)(ncol0 + c) * ld + k0 + half * 16);
}

__device__ __forceinline__ float redmax16(float v) {
    v = fmaxf(v, __shfl_xor(v, 1, 32));
    v = fmaxf(v, __shfl_xor(v, 2, 32));
    v = fmaxf(v, __shfl_xor(v, 4, 32));
    v = fmaxf(v, __shfl_xor(v, 8, 32));
    return v;
}
__device__ __forceinline__ float redsum16(float v) {
    v += __shfl_xor(v, 1, 32);
    v += __shfl_xor(v, 2, 32);
    v += __shfl_xor(v, 4, 32);
    v += __shfl_xor(v, 8, 32);
    return v;
}

// ---------------------------------------------------------------------------
// f32 -> bf16 conversion, 4 elems / thread
// ---------------------------------------------------------------------------
__global__ void __launch_bounds__(256)
cvt_f32_bf16(const float* __restrict__ in, bf16* __restrict__ out, int n4) {
    int i = blockIdx.x * 256 + threadIdx.x;
    if (i >= n4) return;
    float4 v = ((const float4*)in)[i];
    v4bf o;
    o[0] = to_bf16(v.x); o[1] = to_bf16(v.y);
    o[2] = to_bf16(v.z); o[3] = to_bf16(v.w);
    ((v4bf*)out)[i] = o;
}

// ---------------------------------------------------------------------------
// GEMM: Y[m,n] = sum_k X[m,k] * W[n,k]      (X:[M,K] bf16, W:[N,K] bf16)
// Block = 8 waves = 128 rows x 64 cols; W k-slices (64n x 64k) staged in LDS
// via async global->LDS copies, double buffered, shared by all 8 waves.
// mode 0: bf16 out, head-split layout  [B,H,S,dk]   (Q, K)
// mode 1: bf16 out, transposed heads   [B,H,dk,S]   (V^T)
// mode 2: f32  out, row-major [M,N] + bias          (final projection)
// ---------------------------------------------------------------------------
__global__ void __launch_bounds__(256)
wmma_gemm_bf16(const bf16* __restrict__ X, const bf16* __restrict__ W,
               void* __restrict__ out, const float* __restrict__ bias,
               int M, int N, int K, int mode) {
    __shared__ bf16 lds_b[2][64 * 64];        // double-buffered W tile (16 KB)

    int tid  = threadIdx.x;
    int lane = tid & 31;
    int wave = tid >> 5;
    int m0 = blockIdx.y * 128 + wave * 16;
    int n0 = blockIdx.x * 64;

    // staging coords: 64 n-rows x 4 chunks of 16 k-elems (32B) per row
    int st_row = tid >> 2, st_chk = (tid & 3) * 16;
    const bf16* wsrc = W + (size_t)(n0 + st_row) * K + st_chk;

    const int NS = K / 64;                    // 16 stages
    async_cp32(wsrc, &lds_b[0][st_row * 64 + st_chk]);

    v8f acc[4] = {};
    for (int s = 0; s < NS; ++s) {
        int cur = s & 1;
        if (s + 1 < NS) {
            async_cp32(wsrc + (s + 1) * 64, &lds_b[cur ^ 1][st_row * 64 + st_chk]);
            WAIT_ASYNC(2);
        } else {
            WAIT_ASYNC(0);
        }
        __syncthreads();                      // tile [cur] visible to all waves

        const bf16* bt = lds_b[cur];
#pragma unroll
        for (int kk = 0; kk < 64; kk += 32) {
            v16bf a = load_a_frag(X, K, m0, s * 64 + kk, lane);
#pragma unroll
            for (int t = 0; t < 4; ++t) {
                v16bf b = load_b_frag(bt, 64, t * 16, kk, lane);
                acc[t] = wmma_bf16(a, b, acc[t]);
            }
        }
        __syncthreads();                      // done reading before overwrite
    }

    int half = lane >> 4, c = lane & 15;
#pragma unroll
    for (int t = 0; t < 4; ++t) {
        int n = n0 + t * 16 + c;
#pragma unroll
        for (int i = 0; i < 8; ++i) {
            int m = m0 + i + 8 * half;
            float v = acc[t][i];
            if (mode == 2) {
                ((float*)out)[(size_t)m * N + n] = v + bias[n];
            } else {
                int bb = m >> 11, sIdx = m & (SEQ - 1);   // m = bb*SEQ + s
                int h  = n >> 6,  d    = n & (DK - 1);    // n = h*DK + d
                size_t idx = (mode == 0)
                    ? ((size_t)(bb * NHEAD + h) * SEQ + sIdx) * DK + d
                    : ((size_t)(bb * NHEAD + h) * DK + d) * SEQ + sIdx;
                ((bf16*)out)[idx] = to_bf16(v);
            }
        }
    }
}

// ---------------------------------------------------------------------------
// Flash attention: block = 8 waves = 8 adjacent 16-query tiles of ONE (b,h).
// K/V 32-key tiles staged in LDS (async, double buffered), shared by 8 waves.
// Q,K: [B,H,S,dk] bf16.  Vt: [B,H,dk,S] bf16.  ctx out: [B,S,D] bf16.
// ---------------------------------------------------------------------------
__global__ void __launch_bounds__(256)
flash_attn_bf16(const bf16* __restrict__ Q, const bf16* __restrict__ Kmat,
                const bf16* __restrict__ Vt, bf16* __restrict__ ctx) {
    __shared__ bf16 lds_k[2][32 * 64];        // K tile  [key][d]   (2x4 KB)
    __shared__ bf16 lds_v[2][64 * 32];        // V tile  [d][key]   (2x4 KB)
    __shared__ bf16 lds_p[8][16 * 32];        // per-wave P tiles   (8 KB)

    int tid  = threadIdx.x;
    int lane = tid & 31;
    int wave = tid >> 5;
    int bh   = blockIdx.x >> 4;               // 0..31  (b*16 + h)
    int qt   = (blockIdx.x & 15) * 8 + wave;  // 0..127 query tile in (b,h)
    int batch = bh >> 4, h = bh & 15;

    const bf16* Qb = Q    + (size_t)bh * SEQ * DK;
    const bf16* Kb = Kmat + (size_t)bh * SEQ * DK;
    const bf16* Vb = Vt   + (size_t)bh * DK * SEQ;
    bf16* pbuf = lds_p[wave];

    int q0 = qt * 16;
    int half = lane >> 4, c = lane & 15;

    // staging coords: K tile: 32 keys x 8 chunks of 8 elems (16B)
    int kst_row = tid >> 3, kst_chk = (tid & 7) * 8;
    const bf16* ksrc = Kb + (size_t)kst_row * DK + kst_chk;
    // V tile: 64 d-rows x 4 chunks of 8 keys (16B)
    int vst_row = tid >> 2, vst_chk = (tid & 3) * 8;
    const bf16* vsrc = Vb + (size_t)vst_row * SEQ + vst_chk;

    // Q fragments held in registers for the whole key loop (dk=64 -> 2 chunks)
    v16bf aQ0 = load_a_frag(Qb, DK, q0, 0, lane);
    v16bf aQ1 = load_a_frag(Qb, DK, q0, 32, lane);

    float mrun[8], lrun[8];
    v8f o[4] = {};
#pragma unroll
    for (int i = 0; i < 8; ++i) { mrun[i] = -1e30f; lrun[i] = 0.0f; }

    const int NKB = SEQ / 32;                 // 64 key blocks
    async_cp16(ksrc, &lds_k[0][kst_row * 64 + kst_chk]);
    async_cp16(vsrc, &lds_v[0][vst_row * 32 + vst_chk]);

    for (int kbi = 0; kbi < NKB; ++kbi) {
        int cur = kbi & 1;
        if (kbi + 1 < NKB) {
            async_cp16(ksrc + (size_t)(kbi + 1) * 32 * DK,
                       &lds_k[cur ^ 1][kst_row * 64 + kst_chk]);
            async_cp16(vsrc + (kbi + 1) * 32,
                       &lds_v[cur ^ 1][vst_row * 32 + vst_chk]);
            WAIT_ASYNC(2);
        } else {
            WAIT_ASYNC(0);
        }
        __syncthreads();                      // K/V tiles visible

        // ---- scores: 16 queries x 32 keys, from LDS K tile ----
        const bf16* kt = lds_k[cur];
        v8f sc[2] = {};
#pragma unroll
        for (int t = 0; t < 2; ++t) {
            v16bf b0 = load_b_frag(kt, 64, t * 16, 0, lane);
            sc[t] = wmma_bf16(aQ0, b0, sc[t]);
            v16bf b1 = load_b_frag(kt, 64, t * 16, 32, lane);
            sc[t] = wmma_bf16(aQ1, b1, sc[t]);
        }

        // ---- online softmax (row i+8*half lives in this half-wave) ----
#pragma unroll
        for (int i = 0; i < 8; ++i) {
            float s0 = sc[0][i] * 0.125f;     // 1/sqrt(dk)
            float s1 = sc[1][i] * 0.125f;
            float bm = redmax16(fmaxf(s0, s1));
            float mnew = fmaxf(mrun[i], bm);
            float corr = __expf(mrun[i] - mnew);
            float p0 = __expf(s0 - mnew);
            float p1 = __expf(s1 - mnew);
            lrun[i] = lrun[i] * corr + redsum16(p0 + p1);
            mrun[i] = mnew;
#pragma unroll
            for (int j = 0; j < 4; ++j) o[j][i] *= corr;
            int row = i + 8 * half;
            pbuf[row * 32 + c]      = to_bf16(p0);
            pbuf[row * 32 + 16 + c] = to_bf16(p1);
        }
        // accumulator-layout -> A-fragment transpose through per-wave LDS;
        // fence so the compiler cannot reorder cross-lane DS traffic.
        asm volatile("s_wait_dscnt 0" ::: "memory");
        v16bf aP = load_a_frag(pbuf, 32, 0, 0, lane);
        asm volatile("s_wait_dscnt 0" ::: "memory");

        // ---- context accumulate: O += P(16x32) x V(32x64), V from LDS ----
        const bf16* vt = lds_v[cur];
#pragma unroll
        for (int j = 0; j < 4; ++j) {
            v16bf bV = load_b_frag(vt, 32, j * 16, 0, lane);
            o[j] = wmma_bf16(aP, bV, o[j]);
        }
        __syncthreads();                      // done reading before overwrite
    }

    // ---- normalize + store context in [B,S,D] row-major ----
#pragma unroll
    for (int i = 0; i < 8; ++i) {
        float inv = 1.0f / lrun[i];
        int srow = q0 + i + 8 * half;
#pragma unroll
        for (int j = 0; j < 4; ++j) {
            int d = j * 16 + c;
            ctx[((size_t)batch * SEQ + srow) * DMODEL + h * DK + d] =
                to_bf16(o[j][i] * inv);
        }
    }
}

// ---------------------------------------------------------------------------
// Launch
// ---------------------------------------------------------------------------
extern "C" void kernel_launch(void* const* d_in, const int* in_sizes, int n_in,
                              void* d_out, int out_size, void* d_ws, size_t ws_size,
                              hipStream_t stream) {
    (void)in_sizes; (void)n_in; (void)out_size; (void)ws_size;

    const float* x   = (const float*)d_in[0];
    const float* w_q = (const float*)d_in[1];
    const float* w_k = (const float*)d_in[2];
    const float* w_v = (const float*)d_in[3];
    const float* w_o = (const float*)d_in[4];
    const float* b_o = (const float*)d_in[5];

    const size_t XN = (size_t)BATCH * SEQ * DMODEL;   // 4,194,304
    const size_t WN = (size_t)DMODEL * DMODEL;        // 1,048,576

    bf16* ws  = (bf16*)d_ws;
    bf16* xb  = ws;                 // [B*S, D]   (reused for ctx later)
    bf16* wqb = ws + XN;
    bf16* wkb = wqb + WN;
    bf16* wvb = wkb + WN;
    bf16* wob = wvb + WN;
    bf16* Qb  = wob + WN;           // [B,H,S,dk]
    bf16* Kb  = Qb + XN;            // [B,H,S,dk]
    bf16* Vtb = Kb + XN;            // [B,H,dk,S]
    bf16* ctx = xb;                 // alias: x no longer needed post-QKV

    // 1) convert inputs to bf16
    cvt_f32_bf16<<<(int)(XN / 4 / 256), 256, 0, stream>>>(x,   xb,  (int)(XN / 4));
    cvt_f32_bf16<<<(int)(WN / 4 / 256), 256, 0, stream>>>(w_q, wqb, (int)(WN / 4));
    cvt_f32_bf16<<<(int)(WN / 4 / 256), 256, 0, stream>>>(w_k, wkb, (int)(WN / 4));
    cvt_f32_bf16<<<(int)(WN / 4 / 256), 256, 0, stream>>>(w_v, wvb, (int)(WN / 4));
    cvt_f32_bf16<<<(int)(WN / 4 / 256), 256, 0, stream>>>(w_o, wob, (int)(WN / 4));

    // 2) QKV projections (M=4096, N=K=1024)
    const int M = BATCH * SEQ;
    dim3 ggrid(DMODEL / 64, M / 128), gblk(256);
    wmma_gemm_bf16<<<ggrid, gblk, 0, stream>>>(xb, wqb, Qb,  nullptr, M, DMODEL, DMODEL, 0);
    wmma_gemm_bf16<<<ggrid, gblk, 0, stream>>>(xb, wkb, Kb,  nullptr, M, DMODEL, DMODEL, 0);
    wmma_gemm_bf16<<<ggrid, gblk, 0, stream>>>(xb, wvb, Vtb, nullptr, M, DMODEL, DMODEL, 1);

    // 3) attention: 32 (b,h) pairs x 16 blocks; 8 q-tiles per block
    flash_attn_bf16<<<512, 256, 0, stream>>>(Qb, Kb, Vtb, ctx);

    // 4) output projection + bias, f32 out
    wmma_gemm_bf16<<<ggrid, gblk, 0, stream>>>(ctx, wob, d_out, b_o, M, DMODEL, DMODEL, 2);
}
